// Graphormer_47115791237945
// MI455X (gfx1250) — matrix operand order repaired and checked by reading, not compiled, not run on hardware
//
#include <hip/hip_runtime.h>
#include <hip/hip_bf16.h>
#include <cstdint>
#include <cstddef>

typedef __attribute__((ext_vector_type(16))) _Float16 v16h;
typedef __attribute__((ext_vector_type(8)))  _Float16 v8h;
typedef __attribute__((ext_vector_type(8)))  float    v8f;
typedef __attribute__((ext_vector_type(4)))  unsigned int u32x4;
typedef __attribute__((ext_vector_type(8)))  int i32x8;
typedef __attribute__((ext_vector_type(4)))  int i32x4;

// Problem constants
constexpr int cB = 32, cN = 64, cD = 512, cH = 32, cL = 12, cFFN = 512;
constexpr int cS = 65;                 // N + 1
constexpr int cM = cB * cS;            // 2080 real activation rows
constexpr int cMP = 2112;              // padded to a multiple of 64 for 64-row block tiles
constexpr int cDK = 16;
constexpr int cLIN = 369, cLINP = 384; // padded head N

#if defined(__AMDGCN__) && __has_builtin(__builtin_amdgcn_tensor_load_to_lds)
#define HAVE_TDM 1
#else
#define HAVE_TDM 0
#endif

// ---------------------------------------------------------------------------
// TDM: DMA a 64-row x 32-element f16 tile (row stride 512 elems) into LDS.
// D# packing per CDNA5 ISA (group0 + group1, 2D tile).
// This toolchain exposes the 6-arg builtin:
//   (u32x4 g0, i32x8 g1, i32x4 g2, i32x4 g3, i32x8 extra, i32 cpol)
// ---------------------------------------------------------------------------
#if HAVE_TDM
__device__ __forceinline__ void tdm_load_tile_64x32(const _Float16* gsrc,
                                                    unsigned lds_byte_off) {
  unsigned long long ga = (unsigned long long)(uintptr_t)gsrc;
  u32x4 g0;
  g0[0] = 1u;                                   // count=1, user mode, no gather
  g0[1] = lds_byte_off;                         // lds_addr
  g0[2] = (unsigned)ga;                         // global_addr[31:0]
  g0[3] = (unsigned)((ga >> 32) & 0x01ffffffu)  // global_addr[56:32]
          | 0x80000000u;                        // type=2 ("image")
  i32x8 g1;
  g1[0] = 0x10000;                              // data_size=1 -> 2 bytes/elem
  g1[1] = (int)(512u << 16);                    // tensor_dim0 = 512 (lo16 @ 63:48)
  g1[2] = (int)(64u << 16);                     // dim0 hi=0 | tensor_dim1=64
  g1[3] = (int)(32u << 16);                     // dim1 hi=0 | tile_dim0=32
  g1[4] = 64;                                   // tile_dim1=64 | tile_dim2=0
  g1[5] = 512;                                  // tensor_dim0_stride = 512
  g1[6] = 0;                                    // stride0 hi | stride1 lo
  g1[7] = 0;                                    // stride1 hi
  i32x4 z4 = {};
  i32x8 z8 = {};
  __builtin_amdgcn_tensor_load_to_lds(g0, g1, z4, z4, z8, 0);
}
#endif

// ---------------------------------------------------------------------------
// WMMA fragment builders from a 32-elem-pitch f16 tile (LDS or global).
// A: 16x32 (MxK): lanes 0-15 M=lane, elems 0-7 -> K0-7, 8-15 -> K16-23;
//                 lanes 16-31 -> K8-15 / K24-31.
// ---------------------------------------------------------------------------
__device__ __forceinline__ v16h frag_a_p32(const _Float16* base, int lane) {
  const _Float16* p = base + (lane & 15) * 32 + ((lane >> 4) << 3);
  v8h lo = *(const v8h*)p;
  v8h hi = *(const v8h*)(p + 16);
  v16h a;
#pragma unroll
  for (int i = 0; i < 8; ++i) { a[i] = lo[i]; a[8 + i] = hi[i]; }
  return a;
}

// B: 32x16 (KxN) from transposed [N][K] tile: lanes 0-15 N=lane K0-7 / K16-23,
// lanes 16-31 N=lane-16 K8-15 / K24-31 (same K order as A).
__device__ __forceinline__ v16h frag_b_p32(const _Float16* base, int lane) {
  const _Float16* p = base + (lane & 15) * 32 + ((lane >> 4) << 3);
  v8h lo = *(const v8h*)p;
  v8h hi = *(const v8h*)(p + 16);
  v16h b;
#pragma unroll
  for (int i = 0; i < 8; ++i) { b[i] = lo[i]; b[8 + i] = hi[i]; }
  return b;
}

__device__ __forceinline__ float apply_epilogue(float val, int flags) {
  if (flags & 1) val = 0.5f * val * (1.0f + erff(val * 0.70710678118f));
  return val;
}

// ---------------------------------------------------------------------------
// Main GEMM (K fixed = 512): block = 4 waves computing a 64x64 tile.
// A[MP,512] f16 row-major, BT[N,512] f16 (B transposed). Per k-step the
// 64x32 A and B tiles are DMA'd into double-buffered LDS by the TDM
// (tensor_load_to_lds + s_wait_tensorcnt), then each wave runs 4 WMMAs.
// flags: 1 = exact GELU, 2 = accumulate into outf, 4 = store f16 to outh
// ---------------------------------------------------------------------------
__global__ __launch_bounds__(128) void wmma_gemm_tdm_kernel(
    const _Float16* __restrict__ A, const _Float16* __restrict__ BT,
    const float* __restrict__ bias, float* __restrict__ outf,
    _Float16* __restrict__ outh, int N, int flags) {
  constexpr int KSTEPS = 16; // 512 / 32
  __shared__ __align__(16) _Float16 atile[2][64 * 32];
  __shared__ __align__(16) _Float16 btile[2][64 * 32];
  int lane = threadIdx.x & 31;
  int wave = threadIdx.x >> 5;
  int m0 = blockIdx.y << 6;
  int n0 = blockIdx.x << 6;
  const _Float16* Ab = A + (size_t)m0 * 512;
  const _Float16* Bb = BT + (size_t)n0 * 512;

#if HAVE_TDM
  if (threadIdx.x < 32) {
    tdm_load_tile_64x32(Ab, (unsigned)(size_t)&atile[0][0]);
    tdm_load_tile_64x32(Bb, (unsigned)(size_t)&btile[0][0]);
  }
#endif

  v8f acc0 = {}, acc1 = {}, acc2 = {}, acc3 = {};
  for (int s = 0; s < KSTEPS; ++s) {
    int cur = s & 1;
#if HAVE_TDM
    if (threadIdx.x < 32) {
      if (s + 1 < KSTEPS) {
        tdm_load_tile_64x32(Ab + (s + 1) * 32, (unsigned)(size_t)&atile[cur ^ 1][0]);
        tdm_load_tile_64x32(Bb + (s + 1) * 32, (unsigned)(size_t)&btile[cur ^ 1][0]);
        __builtin_amdgcn_s_wait_tensorcnt((short)2); // step-s pair complete
      } else {
        __builtin_amdgcn_s_wait_tensorcnt((short)0);
      }
    }
#else
    {
      int k0 = s * 32;
      for (int idx = threadIdx.x; idx < 64 * 4; idx += 128) {
        int rr = idx >> 2, cc = (idx & 3) * 8;
        *(v8h*)&atile[cur][rr * 32 + cc] =
            *(const v8h*)(Ab + (size_t)rr * 512 + k0 + cc);
        *(v8h*)&btile[cur][rr * 32 + cc] =
            *(const v8h*)(Bb + (size_t)rr * 512 + k0 + cc);
      }
    }
#endif
    __syncthreads(); // staged tiles visible to all 4 waves
    v16h a  = frag_a_p32(&atile[cur][wave * 16 * 32], lane);
    v16h b0 = frag_b_p32(&btile[cur][0 * 16 * 32], lane);
    v16h b1 = frag_b_p32(&btile[cur][1 * 16 * 32], lane);
    v16h b2 = frag_b_p32(&btile[cur][2 * 16 * 32], lane);
    v16h b3 = frag_b_p32(&btile[cur][3 * 16 * 32], lane);
    acc0 = __builtin_amdgcn_wmma_f32_16x16x32_f16(false, a, false, b0, (short)0, acc0, false, false);
    acc1 = __builtin_amdgcn_wmma_f32_16x16x32_f16(false, a, false, b1, (short)0, acc1, false, false);
    acc2 = __builtin_amdgcn_wmma_f32_16x16x32_f16(false, a, false, b2, (short)0, acc2, false, false);
    acc3 = __builtin_amdgcn_wmma_f32_16x16x32_f16(false, a, false, b3, (short)0, acc3, false, false);
    __syncthreads(); // all waves done with buf[cur] before it is refilled
  }

  int rbase = m0 + wave * 16 + ((lane >> 4) << 3);
  int nc = n0 + (lane & 15);
#pragma unroll
  for (int r = 0; r < 8; ++r) {
    int row = rbase + r;
    float vr[4] = {acc0[r], acc1[r], acc2[r], acc3[r]};
#pragma unroll
    for (int t = 0; t < 4; ++t) {
      int n = nc + (t << 4);
      if (n < N) {
        float val = apply_epilogue(vr[t] + bias[n], flags);
        size_t idx = (size_t)row * N + n;
        if (flags & 2)      outf[idx] += val;
        else if (flags & 4) outh[idx] = (_Float16)val;
        else                outf[idx] = val;
      }
    }
  }
}

// ---------------------------------------------------------------------------
// Small direct-load WMMA GEMM (used for the VAE head, M=32): one wave per
// block, 16x64 tile, A/B fetched straight from global (fits in L2 anyway).
// ---------------------------------------------------------------------------
__global__ __launch_bounds__(32) void wmma_gemm_kernel(
    const _Float16* __restrict__ A, const _Float16* __restrict__ BT,
    const float* __restrict__ bias, float* __restrict__ outf,
    _Float16* __restrict__ outh, int M, int N, int K, int flags) {
  int lane = threadIdx.x & 31;
  int m0 = blockIdx.y << 4;
  int n0 = blockIdx.x << 6;
  v8f acc0 = {}, acc1 = {}, acc2 = {}, acc3 = {};
  for (int k0 = 0; k0 < K; k0 += 32) {
    v16h a;
    {
      const _Float16* p = A + (size_t)(m0 + (lane & 15)) * K + k0 + ((lane >> 4) << 3);
      v8h lo = *(const v8h*)p;
      v8h hi = *(const v8h*)(p + 16);
#pragma unroll
      for (int i = 0; i < 8; ++i) { a[i] = lo[i]; a[8 + i] = hi[i]; }
    }
    auto bfrag = [&](int nt) {
      const _Float16* p = BT + (size_t)(n0 + nt * 16 + (lane & 15)) * K + k0 + ((lane >> 4) << 3);
      v8h lo = *(const v8h*)p;
      v8h hi = *(const v8h*)(p + 16);
      v16h b;
#pragma unroll
      for (int i = 0; i < 8; ++i) { b[i] = lo[i]; b[8 + i] = hi[i]; }
      return b;
    };
    v16h b0 = bfrag(0), b1 = bfrag(1), b2 = bfrag(2), b3 = bfrag(3);
    acc0 = __builtin_amdgcn_wmma_f32_16x16x32_f16(false, a, false, b0, (short)0, acc0, false, false);
    acc1 = __builtin_amdgcn_wmma_f32_16x16x32_f16(false, a, false, b1, (short)0, acc1, false, false);
    acc2 = __builtin_amdgcn_wmma_f32_16x16x32_f16(false, a, false, b2, (short)0, acc2, false, false);
    acc3 = __builtin_amdgcn_wmma_f32_16x16x32_f16(false, a, false, b3, (short)0, acc3, false, false);
  }
  int rbase = m0 + ((lane >> 4) << 3);
  int nc = n0 + (lane & 15);
#pragma unroll
  for (int r = 0; r < 8; ++r) {
    int row = rbase + r;
    float vr[4] = {acc0[r], acc1[r], acc2[r], acc3[r]};
#pragma unroll
    for (int t = 0; t < 4; ++t) {
      int n = nc + (t << 4);
      if (n < N) {
        float val = apply_epilogue(vr[t] + bias[n], flags);
        size_t idx = (size_t)row * N + n;
        if (flags & 2)      outf[idx] += val;
        else if (flags & 4) outh[idx] = (_Float16)val;
        else                outf[idx] = val;
      }
    }
  }
}

// ---------------------------------------------------------------------------
// Weight convert + transpose: src f32 [count][K][N] -> dst f16 [count][Npad][K]
// ---------------------------------------------------------------------------
__global__ void convT_kernel(const float* __restrict__ src, _Float16* __restrict__ dst,
                             int K, int N, int Npad, int count) {
  int i = blockIdx.x * blockDim.x + threadIdx.x;
  int total = count * Npad * K;
  if (i >= total) return;
  int k = i % K;
  int r = i / K;
  int n = r % Npad;
  int l = r / Npad;
  float v = (n < N) ? src[((size_t)l * K + k) * N + n] : 0.0f;
  dst[i] = (_Float16)v;
}

// ---------------------------------------------------------------------------
// Structural bias: interior (i>=1, j>=1) of gab[B][H][S][S]
// ---------------------------------------------------------------------------
__global__ __launch_bounds__(32) void edge_bias_kernel(
    const int* __restrict__ rel_pos, const int* __restrict__ edge_input,
    const float* __restrict__ ab, const float* __restrict__ edge_emb,
    const float* __restrict__ edge_dis_emb, const float* __restrict__ rel_pos_emb,
    float* __restrict__ gab) {
  int bid = blockIdx.x;
  int m = bid & 63;
  int r = bid >> 6;
  int n = r & 63;
  int b = r >> 6;
  int lane = threadIdx.x;
  __shared__ float es[5][32];
  const int* ei = edge_input + ((size_t)(b * cN + n) * cN + m) * 5 * 3;
#pragma unroll
  for (int d = 0; d < 5; ++d) {
    float s = 0.f;
#pragma unroll
    for (int f = 0; f < 3; ++f) s += edge_emb[ei[d * 3 + f] * cH + lane];
    es[d][lane] = s * (1.0f / 3.0f);
  }
  __syncthreads();
  int hh = lane;
  float acc = 0.f;
#pragma unroll
  for (int d = 0; d < 5; ++d)
    for (int h2 = 0; h2 < 32; ++h2)
      acc += es[d][h2] * edge_dis_emb[(d * cH + h2) * cH + hh];
  int rp = rel_pos[(size_t)(b * cN + n) * cN + m];
  int rel = (rp == 0) ? 1 : rp;
  if (rel > 1) rel -= 1;
  if (rel > 5) rel = 5;
  float val = rel_pos_emb[rp * cH + hh] + acc / (float)rel;
  int i = n + 1, j = m + 1;
  gab[((size_t)(b * cH + hh)) * cS * cS + i * cS + j] =
      2.0f * ab[(size_t)b * cS * cS + i * cS + j] + val;
}

__global__ void bias_border_kernel(const float* __restrict__ ab,
                                   const float* __restrict__ virt,
                                   float* __restrict__ gab) {
  int i = blockIdx.x * blockDim.x + threadIdx.x;
  int total = cB * cH * cS;
  if (i >= total) return;
  int idx = i % cS;
  int r = i / cS;
  int hh = r & 31;
  int b = r >> 5;
  float t = virt[hh];
  const float* abb = ab + (size_t)b * cS * cS;
  float* g = gab + ((size_t)(b * cH + hh)) * cS * cS;
  g[idx] = 2.0f * abb[idx] + t;
  if (idx > 0) g[idx * cS] = 2.0f * abb[idx * cS] + t;
}

// ---------------------------------------------------------------------------
// Node features -> h[MP][D] (pad rows zeroed so residual accumulation stays
// deterministic)
// ---------------------------------------------------------------------------
__global__ void node_kernel(const int* __restrict__ x, const int* __restrict__ indeg,
                            const float* __restrict__ atom_emb,
                            const float* __restrict__ in_deg_emb,
                            const float* __restrict__ out_deg_emb,
                            const float* __restrict__ graph_token,
                            float* __restrict__ h) {
  int i = blockIdx.x * blockDim.x + threadIdx.x;
  int total = cMP * cD;
  if (i >= total) return;
  int dd = i & (cD - 1);
  int row = i >> 9;
  if (row >= cM) { h[i] = 0.0f; return; }
  int s = row % cS;
  int b = row / cS;
  float val;
  if (s == 0) {
    val = graph_token[dd];
  } else {
    int n = s - 1;
    val = 0.f;
#pragma unroll
    for (int f = 0; f < 3; ++f)
      val += atom_emb[(size_t)x[(b * cN + n) * 3 + f] * cD + dd];
    int dg = indeg[b * cN + n];
    val += in_deg_emb[(size_t)dg * cD + dd] + out_deg_emb[(size_t)dg * cD + dd];
  }
  h[i] = val;
}

// ---------------------------------------------------------------------------
// LayerNorm over D=512; optional f16 and/or f32 outputs
// ---------------------------------------------------------------------------
__global__ __launch_bounds__(256) void ln_kernel(
    const float* __restrict__ x, const float* __restrict__ g,
    const float* __restrict__ bb, _Float16* __restrict__ o16,
    float* __restrict__ o32) {
  int row = blockIdx.x, tid = threadIdx.x;
  const float* xr = x + (size_t)row * cD;
  __shared__ float sm[256];
  float x0 = xr[tid], x1 = xr[tid + 256];
  sm[tid] = x0 + x1;
  __syncthreads();
  for (int st = 128; st > 0; st >>= 1) { if (tid < st) sm[tid] += sm[tid + st]; __syncthreads(); }
  float mu = sm[0] * (1.0f / cD);
  __syncthreads();
  float d0 = x0 - mu, d1 = x1 - mu;
  sm[tid] = d0 * d0 + d1 * d1;
  __syncthreads();
  for (int st = 128; st > 0; st >>= 1) { if (tid < st) sm[tid] += sm[tid + st]; __syncthreads(); }
  float inv = rsqrtf(sm[0] * (1.0f / cD) + 1e-5f);
  float y0 = d0 * inv * g[tid] + bb[tid];
  float y1 = d1 * inv * g[tid + 256] + bb[tid + 256];
  if (o16) {
    o16[(size_t)row * cD + tid] = (_Float16)y0;
    o16[(size_t)row * cD + tid + 256] = (_Float16)y1;
  }
  if (o32) {
    o32[(size_t)row * cD + tid] = y0;
    o32[(size_t)row * cD + tid + 256] = y1;
  }
}

// ---------------------------------------------------------------------------
// Attention: one block per (head, batch)
// ---------------------------------------------------------------------------
__global__ __launch_bounds__(128) void attn_kernel(
    const float* __restrict__ q, const float* __restrict__ k,
    const float* __restrict__ v, const float* __restrict__ gab,
    _Float16* __restrict__ o) {
  int hh = blockIdx.x, b = blockIdx.y;
  __shared__ float ks[cS][cDK], vs[cS][cDK];
  __shared__ float sc[cS][cS + 1];
  int tid = threadIdx.x;
  for (int idx = tid; idx < cS * cDK; idx += 128) {
    int s = idx >> 4, d = idx & 15;
    size_t g = (size_t)(b * cS + s) * cD + hh * cDK + d;
    ks[s][d] = k[g];
    vs[s][d] = v[g];
  }
  __syncthreads();
  const float* gp = gab + ((size_t)(b * cH + hh)) * cS * cS;
  if (tid < cS) {
    int i = tid;
    float qr[cDK];
#pragma unroll
    for (int d = 0; d < cDK; ++d)
      qr[d] = q[(size_t)(b * cS + i) * cD + hh * cDK + d] * 0.25f; // DK^-0.5
    float mx = -3.0e38f;
    for (int j = 0; j < cS; ++j) {
      float s = gp[i * cS + j];
#pragma unroll
      for (int d = 0; d < cDK; ++d) s += qr[d] * ks[j][d];
      sc[i][j] = s;
      mx = fmaxf(mx, s);
    }
    float sum = 0.f;
    for (int j = 0; j < cS; ++j) { float e = __expf(sc[i][j] - mx); sc[i][j] = e; sum += e; }
    float inv = 1.0f / sum;
#pragma unroll
    for (int d = 0; d < cDK; ++d) {
      float od = 0.f;
      for (int j = 0; j < cS; ++j) od += sc[i][j] * vs[j][d];
      o[(size_t)(b * cS + i) * cD + hh * cDK + d] = (_Float16)(od * inv);
    }
  }
}

__global__ void pool_kernel(const float* __restrict__ ln32, _Float16* __restrict__ pooled16) {
  int i = blockIdx.x * blockDim.x + threadIdx.x;
  if (i >= cB * cD) return;
  int b = i >> 9, d = i & (cD - 1);
  float s = 0.f;
  for (int si = 0; si < cS; ++si) s += ln32[((size_t)(b * cS + si)) * cD + d];
  pooled16[i] = (_Float16)s;
}

__global__ void hz_kernel(const float* __restrict__ mu, const float* __restrict__ lv,
                          float* __restrict__ hz) {
  int i = blockIdx.x * blockDim.x + threadIdx.x;
  if (i >= cB * cLIN) return;
  uint32_t s = (uint32_t)i * 747796405u + 2891336453u;
  s ^= s >> 17; s *= 0xed5ad4bbu; s ^= s >> 11; s *= 0xac4c1b51u; s ^= s >> 15;
  uint32_t s2 = s * 0x31848babu + 0x68e31da4u;
  s2 ^= s2 >> 13; s2 *= 0x9e3779b9u; s2 ^= s2 >> 16;
  float u1 = ((s >> 8) + 1u) * (1.0f / 16777216.0f);
  float u2 = (s2 >> 8) * (1.0f / 16777216.0f);
  float eps = sqrtf(-2.0f * logf(u1)) * cosf(6.28318530718f * u2);
  hz[i] = eps * expf(0.5f * lv[i]) + mu[i];
}

// ---------------------------------------------------------------------------
extern "C" void kernel_launch(void* const* d_in, const int* in_sizes, int n_in,
                              void* d_out, int out_size, void* d_ws, size_t ws_size,
                              hipStream_t stream) {
  (void)in_sizes; (void)n_in; (void)out_size; (void)ws_size;
  const int*   x        = (const int*)d_in[0];
  const int*   indeg    = (const int*)d_in[1];
  const int*   rel_pos  = (const int*)d_in[2];
  const int*   edge_in  = (const int*)d_in[3];
  const float* ab       = (const float*)d_in[4];
  const float* atom_emb = (const float*)d_in[5];
  const float* edge_emb = (const float*)d_in[6];
  const float* edge_dis = (const float*)d_in[7];
  const float* relp_emb = (const float*)d_in[8];
  const float* indeg_e  = (const float*)d_in[9];
  const float* outdeg_e = (const float*)d_in[10];
  const float* gtok     = (const float*)d_in[11];
  const float* virt     = (const float*)d_in[12];
  const float* ln1_g = (const float*)d_in[13];
  const float* ln1_b = (const float*)d_in[14];
  const float* Wq = (const float*)d_in[15]; const float* bq = (const float*)d_in[16];
  const float* Wk = (const float*)d_in[17]; const float* bk = (const float*)d_in[18];
  const float* Wv = (const float*)d_in[19]; const float* bv = (const float*)d_in[20];
  const float* Wo = (const float*)d_in[21]; const float* bo = (const float*)d_in[22];
  const float* ln2_g = (const float*)d_in[23];
  const float* ln2_b = (const float*)d_in[24];
  const float* W1 = (const float*)d_in[25]; const float* b1 = (const float*)d_in[26];
  const float* W2 = (const float*)d_in[27]; const float* b2 = (const float*)d_in[28];
  const float* fln_g = (const float*)d_in[29];
  const float* fln_b = (const float*)d_in[30];
  const float* Wm  = (const float*)d_in[31]; const float* bm  = (const float*)d_in[32];
  const float* Wlv = (const float*)d_in[33]; const float* blv = (const float*)d_in[34];

  char* ws = (char*)d_ws;
  size_t off = 0;
  auto carve = [&](size_t bytes) {
    void* p = ws + off;
    off += (bytes + 255) & ~(size_t)255;
    return p;
  };
  const size_t wsz = (size_t)cL * cD * cD * sizeof(_Float16);
  _Float16* wq16 = (_Float16*)carve(wsz);
  _Float16* wk16 = (_Float16*)carve(wsz);
  _Float16* wv16 = (_Float16*)carve(wsz);
  _Float16* wo16 = (_Float16*)carve(wsz);
  _Float16* w1t  = (_Float16*)carve(wsz);
  _Float16* w2t  = (_Float16*)carve(wsz);
  _Float16* wm16  = (_Float16*)carve((size_t)cLINP * cD * sizeof(_Float16));
  _Float16* wlv16 = (_Float16*)carve((size_t)cLINP * cD * sizeof(_Float16));
  float*    h    = (float*)carve((size_t)cMP * cD * sizeof(float));
  _Float16* y16  = (_Float16*)carve((size_t)cMP * cD * sizeof(_Float16));
  _Float16* t16  = (_Float16*)carve((size_t)cMP * cD * sizeof(_Float16));
  _Float16* o16  = (_Float16*)carve((size_t)cMP * cD * sizeof(_Float16));
  float*    q32  = (float*)carve((size_t)cMP * cD * sizeof(float));
  float*    k32  = (float*)carve((size_t)cMP * cD * sizeof(float));
  float*    v32  = (float*)carve((size_t)cMP * cD * sizeof(float));
  float*    gab  = (float*)carve((size_t)cB * cH * cS * cS * sizeof(float));
  float*    ln32 = (float*)carve((size_t)cM * cD * sizeof(float));
  _Float16* pooled16 = (_Float16*)carve((size_t)cB * cD * sizeof(_Float16));

  // ---- weight conversion (f32 [K][N] -> f16 [N][K]) ----
  {
    int total = cL * cD * cD;
    int blocks = (total + 255) / 256;
    convT_kernel<<<blocks, 256, 0, stream>>>(Wq, wq16, cD, cD, cD, cL);
    convT_kernel<<<blocks, 256, 0, stream>>>(Wk, wk16, cD, cD, cD, cL);
    convT_kernel<<<blocks, 256, 0, stream>>>(Wv, wv16, cD, cD, cD, cL);
    convT_kernel<<<blocks, 256, 0, stream>>>(Wo, wo16, cD, cD, cD, cL);
    convT_kernel<<<blocks, 256, 0, stream>>>(W1, w1t, cD, cFFN, cFFN, cL);
    convT_kernel<<<blocks, 256, 0, stream>>>(W2, w2t, cFFN, cD, cD, cL);
    int th = cLINP * cD;
    convT_kernel<<<(th + 255) / 256, 256, 0, stream>>>(Wm, wm16, cD, cLIN, cLINP, 1);
    convT_kernel<<<(th + 255) / 256, 256, 0, stream>>>(Wlv, wlv16, cD, cLIN, cLINP, 1);
  }

  // ---- structural attention bias ----
  edge_bias_kernel<<<cB * cN * cN, 32, 0, stream>>>(rel_pos, edge_in, ab, edge_emb,
                                                    edge_dis, relp_emb, gab);
  {
    int total = cB * cH * cS;
    bias_border_kernel<<<(total + 255) / 256, 256, 0, stream>>>(ab, virt, gab);
  }

  // ---- node features ----
  {
    int total = cMP * cD;
    node_kernel<<<(total + 255) / 256, 256, 0, stream>>>(x, indeg, atom_emb, indeg_e,
                                                         outdeg_e, gtok, h);
  }

  dim3 gemmGrid(cD / 64, cMP / 64); // 8 x 33, 128-thread blocks
  for (int i = 0; i < cL; ++i) {
    size_t wofs = (size_t)i * cD * cD;
    ln_kernel<<<cM, 256, 0, stream>>>(h, ln1_g + i * cD, ln1_b + i * cD, y16, nullptr);
    wmma_gemm_tdm_kernel<<<gemmGrid, 128, 0, stream>>>(y16, wq16 + wofs, bq + i * cD,
                                                       q32, nullptr, cD, 0);
    wmma_gemm_tdm_kernel<<<gemmGrid, 128, 0, stream>>>(y16, wk16 + wofs, bk + i * cD,
                                                       k32, nullptr, cD, 0);
    wmma_gemm_tdm_kernel<<<gemmGrid, 128, 0, stream>>>(y16, wv16 + wofs, bv + i * cD,
                                                       v32, nullptr, cD, 0);
    attn_kernel<<<dim3(cH, cB), 128, 0, stream>>>(q32, k32, v32, gab, o16);
    wmma_gemm_tdm_kernel<<<gemmGrid, 128, 0, stream>>>(o16, wo16 + wofs, bo + i * cD,
                                                       h, nullptr, cD, 2);
    ln_kernel<<<cM, 256, 0, stream>>>(h, ln2_g + i * cD, ln2_b + i * cD, y16, nullptr);
    wmma_gemm_tdm_kernel<<<gemmGrid, 128, 0, stream>>>(y16, w1t + wofs, b1 + i * cFFN,
                                                       nullptr, t16, cFFN, 1 | 4);
    wmma_gemm_tdm_kernel<<<gemmGrid, 128, 0, stream>>>(t16, w2t + wofs, b2 + i * cD,
                                                       h, nullptr, cD, 2);
  }

  // ---- final LN, pooling, VAE head ----
  ln_kernel<<<cM, 256, 0, stream>>>(h, fln_g, fln_b, nullptr, ln32);
  pool_kernel<<<(cB * cD + 255) / 256, 256, 0, stream>>>(ln32, pooled16);

  float* out = (float*)d_out;
  float* hz = out;
  float* mu = out + (size_t)cB * cLIN;
  float* lv = out + (size_t)2 * cB * cLIN;
  dim3 headGrid((cLINP + 63) / 64, cB / 16); // 6 x 2
  wmma_gemm_kernel<<<headGrid, 32, 0, stream>>>(pooled16, wm16, bm, mu, nullptr,
                                                cB, cLIN, cD, 0);
  wmma_gemm_kernel<<<headGrid, 32, 0, stream>>>(pooled16, wlv16, blv, lv, nullptr,
                                                cB, cLIN, cD, 0);
  hz_kernel<<<(cB * cLIN + 255) / 256, 256, 0, stream>>>(mu, lv, hz);
}
